// KPlexPool_22454089024244
// MI455X (gfx1250) — compile-verified, matrix operand order checked by reading, not run on hardware
//
#include <hip/hip_runtime.h>
#include <hip/hip_bf16.h>

typedef __attribute__((ext_vector_type(16))) _Float16 v16h;
typedef __attribute__((ext_vector_type(8)))  _Float16 v8h;
typedef __attribute__((ext_vector_type(8)))  float    v8f;

#define N0c 50000
#define E0c 800000
#define N1c 20000
#define E1c 320000
#define Ac  100000
#define Bc  64
#define Cc  10
#define Hc  128

// ---------------- utility kernels ----------------

__global__ void k_f32_to_f16(const float* __restrict__ s, _Float16* __restrict__ d, int n) {
    int i = blockIdx.x * blockDim.x + threadIdx.x;
    if (i < n) d[i] = (_Float16)s[i];
}

__global__ void k_set_f32(float* __restrict__ d, float v, int n) {
    int i = blockIdx.x * blockDim.x + threadIdx.x;
    if (i < n) d[i] = v;
}

__global__ void k_rsqrt_inplace(float* __restrict__ d, int n) {
    int i = blockIdx.x * blockDim.x + threadIdx.x;
    if (i < n) d[i] = rsqrtf(fmaxf(d[i], 1e-12f));
}

__global__ void k_edge_deg(const int* __restrict__ col, const float* __restrict__ ew,
                           float* __restrict__ deg, int E) {
    int e = blockIdx.x * blockDim.x + threadIdx.x;
    if (e < E) atomicAdd(&deg[col[e]], ew[e]);
}

// agg[i,:] = dinv[i]^2 * xw[i,:]   (self-loop term; also initializes agg)
__global__ void k_selfloop_init(const float* __restrict__ dinv, const float* __restrict__ xw,
                                float* __restrict__ agg, int n) {
    int i = blockIdx.x * blockDim.x + threadIdx.x;
    if (i < n * Hc) {
        int r = i >> 7;
        float di = dinv[r];
        agg[i] = di * di * xw[i];
    }
}

// one wave per edge: agg[col,:] += dinv[row]*ew*dinv[col] * xw[row,:]
__global__ void k_edge_scatter(const int* __restrict__ row, const int* __restrict__ col,
                               const float* __restrict__ ew, const float* __restrict__ dinv,
                               const float* __restrict__ xw, float* __restrict__ agg, int E) {
    int t = blockIdx.x * 256 + threadIdx.x;
    int e = t >> 5;
    if (e >= E) return;
    int lane = t & 31;
    int r = row[e], c = col[e];
    float coef = dinv[r] * ew[e] * dinv[c];
    float4 v = ((const float4*)(xw + (size_t)r * Hc))[lane];
    float* dst = agg + (size_t)c * Hc + lane * 4;
    atomicAdd(dst + 0, coef * v.x);
    atomicAdd(dst + 1, coef * v.y);
    atomicAdd(dst + 2, coef * v.z);
    atomicAdd(dst + 3, coef * v.w);
}

// out_h[r, colofs+j] = f16(relu(agg[r,j] + bias[j]))
__global__ void k_bias_relu_f16(const float* __restrict__ agg, const float* __restrict__ bias,
                                _Float16* __restrict__ outh, int ldh, int colofs, int n) {
    int i = blockIdx.x * blockDim.x + threadIdx.x;
    if (i < n * Hc) {
        int r = i >> 7, j = i & 127;
        float v = fmaxf(agg[i] + bias[j], 0.0f);
        outh[(size_t)r * ldh + colofs + j] = (_Float16)v;
    }
}

// ---------------- weight pack: [K x 128] f32 -> WMMA B-fragment order f16 ----------------
// packed[(((ks*8 + ntile)*32 + lane)*16) + e] with e=2j+par mapping to the ISA 16-bit
// B 32x16 lane layout: k = ks*32 + koff(j,half) + par, n = ntile*16 + (lane&15).
__global__ void k_pack_w(const float* __restrict__ w, _Float16* __restrict__ p, int K) {
    int i = blockIdx.x * blockDim.x + threadIdx.x;
    if (i >= K * 128) return;
    int e    = i & 15;
    int lane = (i >> 4) & 31;
    int nt   = (i >> 9) & 7;
    int ks   = i >> 12;
    int j = e >> 1, par = e & 1;
    int half = lane >> 4;
    int koff = ((j < 4) ? (2 * j) : (16 + 2 * (j - 4))) + 8 * half;
    int k = ks * 32 + koff + par;
    int n = nt * 16 + (lane & 15);
    p[i] = (_Float16)w[k * 128 + n];
}

// ---------------- WMMA GEMM: out[M x 128] = A[M x K](f16) @ Bp(packed f16) ----------------
// 256 threads = 8 waves; block covers a 16-row stripe, each wave one 16-col N-tile.
// REQUIRES: M % 16 == 0 (true for all call sites: 50000, 20000), lda % 8 == 0.
// K templated (128/256) for full unroll; FUSED adds bias+relu epilogue.
template<int K, int FUSED>
__global__ __launch_bounds__(256)
void k_gemm_wmma(const _Float16* __restrict__ A, int lda,
                 const _Float16* __restrict__ Bp,
                 const float* __restrict__ bias,
                 float* __restrict__ outF, int M) {
    // LDA = K+8 halves: rows stay 16B-aligned (272B / 528B) and bank-rotate by 4 dwords/row
    const int LDA = K + 8;
    __shared__ _Float16 As[16 * LDA];

    int m0 = blockIdx.x * 16;

    // cooperative vectorized stage of the 16 x K A-tile (b128 load -> b128 ds store)
    const int KV = K / 8;                 // 8-half vectors per row
#pragma unroll
    for (int vi = threadIdx.x; vi < 16 * KV; vi += 256) {
        int r = vi / KV, cv = vi - r * KV;
        *(v8h*)(As + r * LDA + cv * 8) =
            *(const v8h*)(A + (size_t)(m0 + r) * lda + cv * 8);
    }
    __syncthreads();

    int lane   = threadIdx.x & 31;
    int half   = lane >> 4;          // 0: lanes 0-15, 1: lanes 16-31
    int lanelo = lane & 15;
    int wave   = threadIdx.x >> 5;
    int ncol0  = wave * 16;

    const v8h* arow = (const v8h*)(As + lanelo * LDA);          // 16B-aligned
    const v16h* bfp = (const v16h*)(Bp + (size_t)((wave * 32) + lane) * 16);

    v8f c = {};
#pragma unroll
    for (int ks = 0; ks < K / 32; ++ks) {
        // A fragment: two contiguous 8-half runs -> 2x ds_load_b128, concat
        v8h lo = arow[ks * 4 + half];
        v8h hi = arow[ks * 4 + 2 + half];
        v16h af = __builtin_shufflevector(lo, hi, 0, 1, 2, 3, 4, 5, 6, 7,
                                          8, 9, 10, 11, 12, 13, 14, 15);
        // B fragment: single contiguous 32B vector load from packed weights
        v16h bf = bfp[(size_t)ks * 8 * 32];
        c = __builtin_amdgcn_wmma_f32_16x16x32_f16(false, af, false, bf,
                                                   (short)0, c, false, false);
    }

    // unconditional epilogue: 8 stores off one base, constant 512B offsets
    int gc = ncol0 + lanelo;
    float* orow = outF + (size_t)(m0 + 8 * half) * 128 + gc;
    float bval = FUSED ? bias[gc] : 0.0f;
#pragma unroll
    for (int v = 0; v < 8; ++v) {
        float val = c[v];                 // C/D layout: VGPR v -> M = v (+8 for hi lanes)
        if (FUSED) val = fmaxf(val + bval, 0.0f);
        orow[(size_t)v * 128] = val;
    }
}

// ---------------- pooling ----------------

// cover pool: one wave per assignment; sum into cols [0,128), max into [128,256)
__global__ void k_cover_pool(const float* __restrict__ h, const int* __restrict__ crow,
                             const int* __restrict__ ccol, float* __restrict__ h2, int nA) {
    int t = blockIdx.x * 256 + threadIdx.x;
    int a = t >> 5;
    if (a >= nA) return;
    int lane = t & 31;
    int r = crow[a], cc = ccol[a];
    float4 v = ((const float4*)(h + (size_t)r * Hc))[lane];
    float* ds = h2 + (size_t)cc * 256 + lane * 4;
    atomicAdd(ds + 0, v.x);
    atomicAdd(ds + 1, v.y);
    atomicAdd(ds + 2, v.z);
    atomicAdd(ds + 3, v.w);
    unsigned* dm = (unsigned*)(h2 + (size_t)cc * 256 + 128 + lane * 4);
    atomicMax(dm + 0, __float_as_uint(v.x));   // valid: values are post-ReLU >= 0
    atomicMax(dm + 1, __float_as_uint(v.y));
    atomicMax(dm + 2, __float_as_uint(v.z));
    atomicMax(dm + 3, __float_as_uint(v.w));
}

// graph-level pool: z[b, sofs+j] += h[i,j]; z[b, mofs+j] = max(...)
__global__ void k_batch_pool(const float* __restrict__ h, const int* __restrict__ batch,
                             float* __restrict__ z, int n, int sofs, int mofs) {
    int i = blockIdx.x * blockDim.x + threadIdx.x;
    if (i >= n * Hc) return;
    int r = i >> 7, j = i & 127;
    int b = batch[r];
    float v = h[i];
    atomicAdd(&z[b * 512 + sofs + j], v);
    atomicMax((unsigned*)&z[b * 512 + mofs + j], __float_as_uint(v));
}

// ---------------- classifier head ----------------

// z1[b,o] = relu( sum_k (z[b,k]*gamma[k]*s + beta[k]) * w[k,o] + bias[o] )
__global__ void k_head1(const float* __restrict__ z, const float* __restrict__ gamma,
                        const float* __restrict__ beta, const float* __restrict__ w,
                        const float* __restrict__ bias, float* __restrict__ z1) {
    int b = blockIdx.x;       // 0..63
    int o = threadIdx.x;      // 0..127
    const float s = rsqrtf(1.0f + 1e-5f);
    float acc = bias[o];
    for (int k = 0; k < 512; ++k) {
        float zk = z[b * 512 + k] * (gamma[k] * s) + beta[k];
        acc += zk * w[k * 128 + o];
    }
    z1[b * 128 + o] = fmaxf(acc, 0.0f);
}

// logits + softmax, one thread per batch row
__global__ void k_head2(const float* __restrict__ z1, const float* __restrict__ w,
                        const float* __restrict__ bias, float* __restrict__ out) {
    int b = blockIdx.x * blockDim.x + threadIdx.x;
    if (b >= Bc) return;
    float lg[Cc];
    float mx = -3.0e38f;
    for (int c = 0; c < Cc; ++c) {
        float acc = bias[c];
        for (int k = 0; k < 128; ++k) acc += z1[b * 128 + k] * w[k * Cc + c];
        lg[c] = acc;
        mx = fmaxf(mx, acc);
    }
    float ssum = 0.0f;
    for (int c = 0; c < Cc; ++c) { lg[c] = __expf(lg[c] - mx); ssum += lg[c]; }
    float inv = 1.0f / ssum;
    for (int c = 0; c < Cc; ++c) out[b * Cc + c] = lg[c] * inv;
}

// ---------------- launch ----------------

extern "C" void kernel_launch(void* const* d_in, const int* in_sizes, int n_in,
                              void* d_out, int out_size, void* d_ws, size_t ws_size,
                              hipStream_t stream) {
    (void)in_sizes; (void)n_in; (void)out_size; (void)ws_size;

    const float* x        = (const float*)d_in[0];
    const int*   ei0      = (const int*)d_in[1];
    const float* ew0      = (const float*)d_in[2];
    const int*   batch0   = (const int*)d_in[3];
    const int*   crow     = (const int*)d_in[4];
    const int*   ccol     = (const int*)d_in[5];
    const int*   ei1      = (const int*)d_in[6];
    const float* ew1      = (const float*)d_in[7];
    const int*   batch1   = (const int*)d_in[8];
    const float* w_in1    = (const float*)d_in[9];
    const float* b_in1    = (const float*)d_in[10];
    const float* w_in2    = (const float*)d_in[11];
    const float* b_in2    = (const float*)d_in[12];
    const float* w_inl    = (const float*)d_in[13];
    const float* b_inl    = (const float*)d_in[14];
    const float* w_b1     = (const float*)d_in[15];
    const float* b_b1     = (const float*)d_in[16];
    const float* w_b2     = (const float*)d_in[17];
    const float* b_b2     = (const float*)d_in[18];
    const float* w_bl     = (const float*)d_in[19];
    const float* b_bl     = (const float*)d_in[20];
    const float* bn_g     = (const float*)d_in[21];
    const float* bn_b     = (const float*)d_in[22];
    const float* w_l1     = (const float*)d_in[23];
    const float* b_l1     = (const float*)d_in[24];
    const float* w_l2     = (const float*)d_in[25];
    const float* b_l2     = (const float*)d_in[26];
    float* out            = (float*)d_out;

    const int* row0 = ei0;            const int* col0 = ei0 + E0c;
    const int* row1 = ei1;            const int* col1 = ei1 + E1c;

    // ---- workspace carve-up (with aliasing) ----
    char* base = (char*)d_ws;
    size_t off = 0;
    auto alloc = [&](size_t bytes) -> char* {
        char* p = base + off;
        off += (bytes + 255) & ~(size_t)255;
        return p;
    };
    _Float16* wp_in1 = (_Float16*)alloc((size_t)128 * 128 * 2);   // packed weights
    _Float16* wp_in2 = (_Float16*)alloc((size_t)128 * 128 * 2);
    _Float16* wp_inl = (_Float16*)alloc((size_t)256 * 128 * 2);
    _Float16* wp_b1  = (_Float16*)alloc((size_t)256 * 128 * 2);
    _Float16* wp_b2  = (_Float16*)alloc((size_t)128 * 128 * 2);
    _Float16* wp_bl  = (_Float16*)alloc((size_t)256 * 128 * 2);
    float*    xw     = (float*)alloc((size_t)N0c * 128 * 4);          // shared xw buffer
    float*    agg    = (float*)alloc((size_t)N0c * 128 * 4);          // aliases h2 [N1 x 256]
    _Float16* cat0   = (_Float16*)alloc((size_t)N0c * 256 * 2);       // aliases h2h + cat1
    float*    h      = (float*)alloc((size_t)N0c * 128 * 4);          // aliases xh16 (f16)
    float*    deg    = (float*)alloc((size_t)N0c * 4);
    float*    hh     = (float*)alloc((size_t)N1c * 128 * 4);
    float*    z      = (float*)alloc((size_t)Bc * 512 * 4);
    float*    z1     = (float*)alloc((size_t)Bc * 128 * 4);

    float*    h2   = agg;                            // N1 x 256 f32 (agg dead then)
    _Float16* xh16 = (_Float16*)h;                   // dead before h is written
    _Float16* h2h  = cat0;                           // cat0 dead after GEMM3
    _Float16* cat1 = cat0 + (size_t)N1c * 256;

    auto cdiv = [](long a, long b) { return (int)((a + b - 1) / b); };

    // ---- pack weights to WMMA fragment order, convert input features ----
    k_pack_w<<<cdiv(128 * 128, 256), 256, 0, stream>>>(w_in1, wp_in1, 128);
    k_pack_w<<<cdiv(128 * 128, 256), 256, 0, stream>>>(w_in2, wp_in2, 128);
    k_pack_w<<<cdiv(256 * 128, 256), 256, 0, stream>>>(w_inl, wp_inl, 256);
    k_pack_w<<<cdiv(256 * 128, 256), 256, 0, stream>>>(w_b1, wp_b1, 256);
    k_pack_w<<<cdiv(128 * 128, 256), 256, 0, stream>>>(w_b2, wp_b2, 128);
    k_pack_w<<<cdiv(256 * 128, 256), 256, 0, stream>>>(w_bl, wp_bl, 256);
    k_f32_to_f16<<<cdiv((long)N0c * 128, 256), 256, 0, stream>>>(x, xh16, N0c * 128);

    // ---- graph 0: degrees ----
    k_set_f32<<<cdiv(N0c, 256), 256, 0, stream>>>(deg, 1.0f, N0c);    // self-loop weight
    k_edge_deg<<<cdiv(E0c, 256), 256, 0, stream>>>(col0, ew0, deg, E0c);
    k_rsqrt_inplace<<<cdiv(N0c, 256), 256, 0, stream>>>(deg, N0c);    // deg -> dinv

    // ---- graph 0, GCN layer 1 ----
    k_gemm_wmma<128, 0><<<cdiv(N0c, 16), 256, 0, stream>>>(xh16, 128, wp_in1, nullptr, xw, N0c);
    k_selfloop_init<<<cdiv((long)N0c * 128, 256), 256, 0, stream>>>(deg, xw, agg, N0c);
    k_edge_scatter<<<cdiv((long)E0c * 32, 256), 256, 0, stream>>>(row0, col0, ew0, deg, xw, agg, E0c);
    k_bias_relu_f16<<<cdiv((long)N0c * 128, 256), 256, 0, stream>>>(agg, b_in1, cat0, 256, 0, N0c);

    // ---- graph 0, GCN layer 2 (input = cat0 cols [0,128)) ----
    k_gemm_wmma<128, 0><<<cdiv(N0c, 16), 256, 0, stream>>>(cat0, 256, wp_in2, nullptr, xw, N0c);
    k_selfloop_init<<<cdiv((long)N0c * 128, 256), 256, 0, stream>>>(deg, xw, agg, N0c);
    k_edge_scatter<<<cdiv((long)E0c * 32, 256), 256, 0, stream>>>(row0, col0, ew0, deg, xw, agg, E0c);
    k_bias_relu_f16<<<cdiv((long)N0c * 128, 256), 256, 0, stream>>>(agg, b_in2, cat0, 256, 128, N0c);

    // ---- jumping-knowledge linear + relu -> h ----
    k_gemm_wmma<256, 1><<<cdiv(N0c, 16), 256, 0, stream>>>(cat0, 256, wp_inl, b_inl, h, N0c);

    // ---- graph-level pool of h into z cols [0,256) ----
    k_set_f32<<<cdiv(Bc * 512, 256), 256, 0, stream>>>(z, 0.0f, Bc * 512);
    k_batch_pool<<<cdiv((long)N0c * 128, 256), 256, 0, stream>>>(h, batch0, z, N0c, 0, 128);

    // ---- cover pool -> h2 [N1 x 256] ----
    k_set_f32<<<cdiv((long)N1c * 256, 256), 256, 0, stream>>>(h2, 0.0f, N1c * 256);
    k_cover_pool<<<cdiv((long)Ac * 32, 256), 256, 0, stream>>>(h, crow, ccol, h2, Ac);
    k_f32_to_f16<<<cdiv((long)N1c * 256, 256), 256, 0, stream>>>(h2, h2h, N1c * 256);

    // ---- graph 1: degrees ----
    k_set_f32<<<cdiv(N1c, 256), 256, 0, stream>>>(deg, 1.0f, N1c);
    k_edge_deg<<<cdiv(E1c, 256), 256, 0, stream>>>(col1, ew1, deg, E1c);
    k_rsqrt_inplace<<<cdiv(N1c, 256), 256, 0, stream>>>(deg, N1c);

    // ---- graph 1, GCN layer 1 (K=256) ----
    k_gemm_wmma<256, 0><<<cdiv(N1c, 16), 256, 0, stream>>>(h2h, 256, wp_b1, nullptr, xw, N1c);
    k_selfloop_init<<<cdiv((long)N1c * 128, 256), 256, 0, stream>>>(deg, xw, agg, N1c);
    k_edge_scatter<<<cdiv((long)E1c * 32, 256), 256, 0, stream>>>(row1, col1, ew1, deg, xw, agg, E1c);
    k_bias_relu_f16<<<cdiv((long)N1c * 128, 256), 256, 0, stream>>>(agg, b_b1, cat1, 256, 0, N1c);

    // ---- graph 1, GCN layer 2 ----
    k_gemm_wmma<128, 0><<<cdiv(N1c, 16), 256, 0, stream>>>(cat1, 256, wp_b2, nullptr, xw, N1c);
    k_selfloop_init<<<cdiv((long)N1c * 128, 256), 256, 0, stream>>>(deg, xw, agg, N1c);
    k_edge_scatter<<<cdiv((long)E1c * 32, 256), 256, 0, stream>>>(row1, col1, ew1, deg, xw, agg, E1c);
    k_bias_relu_f16<<<cdiv((long)N1c * 128, 256), 256, 0, stream>>>(agg, b_b2, cat1, 256, 128, N1c);

    // ---- jumping-knowledge linear + relu -> hh ----
    k_gemm_wmma<256, 1><<<cdiv(N1c, 16), 256, 0, stream>>>(cat1, 256, wp_bl, b_bl, hh, N1c);

    // ---- graph-level pool of hh into z cols [256,512) ----
    k_batch_pool<<<cdiv((long)N1c * 128, 256), 256, 0, stream>>>(hh, batch1, z, N1c, 256, 384);

    // ---- head: BN + fc1 + relu, fc2 + softmax ----
    k_head1<<<Bc, 128, 0, stream>>>(z, bn_g, bn_b, w_l1, b_l1, z1);
    k_head2<<<1, Bc, 0, stream>>>(z1, w_l2, b_l2, out);
}
